// CoReMultiheadAttention_74698071212077
// MI455X (gfx1250) — compile-verified, matrix operand order checked by reading, not run on hardware
//
#include <hip/hip_runtime.h>
#include <hip/hip_bf16.h>
#include <math.h>
#include <stdint.h>

// Problem constants (from reference)
static constexpr int NHEADS = 16;
static constexpr int EMB    = 1024;
static constexpr int NBATCH = 16;
static constexpr int NSEQ   = 512;
static constexpr int HDIM   = 64;     // EMB / NHEADS
static constexpr int NBH    = 256;    // NBATCH * NHEADS

typedef __attribute__((ext_vector_type(16))) _Float16 v16h;
typedef __attribute__((ext_vector_type(8)))  _Float16 v8h;
typedef __attribute__((ext_vector_type(8)))  float    v8f;
typedef __attribute__((ext_vector_type(4)))  unsigned int u32x4;
typedef __attribute__((ext_vector_type(8)))  unsigned int u32x8;

union Frag16 { v16h v; v8h h[2]; };

__device__ __forceinline__ float wave_sum32(float v) {
#pragma unroll
    for (int off = 16; off > 0; off >>= 1)
        v += __shfl_xor(v, off, 32);
    return v;
}

// ---------------------------------------------------------------------------
// Kernel A: packed in-projection.  One block per output column j (0..3071),
// 16 waves per block = one wave per batch row i. q is pre-scaled by 1/sqrt(hd).
// ---------------------------------------------------------------------------
__global__ __launch_bounds__(512)
void inproj_kernel(const float* __restrict__ q_step, const float* __restrict__ k_step,
                   const float* __restrict__ v_step, const float* __restrict__ W,
                   const float* __restrict__ bias,
                   float* __restrict__ q_s, float* __restrict__ k_s,
                   float* __restrict__ v_s) {
    const int j    = blockIdx.x;          // 0..3*EMB-1
    const int wave = threadIdx.x >> 5;    // batch row i, 0..15
    const int lane = threadIdx.x & 31;
    const int sel  = j >> 10;             // 0=q 1=k 2=v
    const float* x  = (sel == 0) ? q_step : (sel == 1) ? k_step : v_step;
    const float* w  = W + (size_t)j * EMB;
    const float* xr = x + (size_t)wave * EMB;
    float acc = 0.f;
    for (int c = lane; c < EMB; c += 32)
        acc += xr[c] * w[c];
    acc = wave_sum32(acc);
    if (lane == 0) {
        float r = acc + bias[j];
        int jj = j & (EMB - 1);
        int hh = jj >> 6;
        int ee = jj & (HDIM - 1);
        int b  = wave * NHEADS + hh;
        float* dst = (sel == 0) ? q_s : (sel == 1) ? k_s : v_s;
        dst[b * HDIM + ee] = (sel == 0) ? r * 0.125f : r;  // 1/sqrt(64)
    }
}

// ---------------------------------------------------------------------------
// Kernel C: convert out_proj_weight f32 -> f16 (row-major, used as WMMA B).
// ---------------------------------------------------------------------------
__global__ void cvt_f16_kernel(const float* __restrict__ W, _Float16* __restrict__ Wh, int n) {
    int i = blockIdx.x * blockDim.x + threadIdx.x;
    for (; i < n; i += gridDim.x * blockDim.x)
        Wh[i] = (_Float16)W[i];
}

// ---------------------------------------------------------------------------
// Kernel B: continual retroactive attention update.  One block (256 thr,
// 8 waves) per b in [0,256).  Writes normalized attention directly into the
// GEMM-ready f16 matrix: attn_h[(nt*NBATCH + nb)*EMB + h*HDIM + e].
// ---------------------------------------------------------------------------
__global__ __launch_bounds__(256)
void attn_update_kernel(const float* __restrict__ d_mem, const float* __restrict__ AV_mem,
                        const float* __restrict__ Q_mem, const float* __restrict__ K_T_mem,
                        const float* __restrict__ V_mem,
                        const float* __restrict__ q_s, const float* __restrict__ k_s,
                        const float* __restrict__ v_s,
                        _Float16* __restrict__ attn_h) {
    __shared__ float kcol0[HDIM], kvec[HDIM], vvec[HDIM], qvec[HDIM], v0row[HDIM];
    __shared__ float abot[NSEQ];
    __shared__ float red[8];
    __shared__ float pp[4][HDIM];
    __shared__ float dbot_s;

    const int b  = blockIdx.x;
    const int t  = threadIdx.x;
    const int nb = b >> 4;        // batch index   (b = nb*NHEADS + h)
    const int hh = b & 15;        // head index
    if (t < HDIM) {
        kcol0[t] = K_T_mem[((size_t)b * HDIM + t) * NSEQ];        // old column 0
        kvec[t]  = k_s[b * HDIM + t];                             // new k
        vvec[t]  = v_s[b * HDIM + t];                             // new v
        qvec[t]  = q_s[b * HDIM + t];                             // new q (scaled)
        v0row[t] = V_mem[(size_t)b * NSEQ * HDIM + t];            // old row 0 of V
    }
    __syncthreads();

    const int wave = t >> 5, lane = t & 31;
    // ---- top rows n = 0..NSEQ-2: rank-1 retroactive update ----
    for (int n = wave; n < NSEQ - 1; n += 8) {
        const size_t base = ((size_t)b * (NSEQ - 1) + n) * HDIM;
        float q1 = Q_mem[base + lane];
        float q2 = Q_mem[base + lane + 32];
        float dl = q1 * kcol0[lane] + q2 * kcol0[lane + 32];
        float dr = q1 * kvec[lane]  + q2 * kvec[lane + 32];
        dl = wave_sum32(dl);
        dr = wave_sum32(dr);
        float a_left  = expf(dl);
        float a_right = expf(dr);
        float dt  = d_mem[(size_t)b * (NSEQ - 1) + n] - a_left + a_right;
        float inv = 1.0f / dt;
#pragma unroll
        for (int half = 0; half < 2; ++half) {
            int e = lane + half * 32;
            float av = AV_mem[base + e] - a_left * v0row[e] + a_right * vvec[e];
            attn_h[((size_t)(n * NBATCH + nb)) * EMB + hh * HDIM + e] = (_Float16)(av * inv);
        }
    }
    __syncthreads();

    // ---- bottom row: full softmax row of the new query over rolled K/V ----
    float lsum = 0.f;
    for (int s = t; s < NSEQ; s += 256) {
        float d = 0.f;
        if (s < NSEQ - 1) {
            const float* kc = K_T_mem + (size_t)b * HDIM * NSEQ + (s + 1);
#pragma unroll 8
            for (int e = 0; e < HDIM; ++e)
                d += qvec[e] * kc[(size_t)e * NSEQ];
        } else {
#pragma unroll 8
            for (int e = 0; e < HDIM; ++e)
                d += qvec[e] * kvec[e];
        }
        float a = expf(d);
        abot[s] = a;
        lsum += a;
    }
    lsum = wave_sum32(lsum);
    if (lane == 0) red[wave] = lsum;
    __syncthreads();
    if (t == 0) {
        float s = 0.f;
#pragma unroll
        for (int i = 0; i < 8; ++i) s += red[i];
        dbot_s = s;
    }
    {
        const int e = t & 63, chunk = t >> 6;
        float part = 0.f;
        for (int s = chunk * 128; s < chunk * 128 + 128; ++s) {
            float vv = (s < NSEQ - 1) ? V_mem[((size_t)b * NSEQ + s + 1) * HDIM + e] : vvec[e];
            part += abot[s] * vv;
        }
        pp[chunk][e] = part;
    }
    __syncthreads();
    if (t < HDIM) {
        float av = pp[0][t] + pp[1][t] + pp[2][t] + pp[3][t];
        attn_h[((size_t)((NSEQ - 1) * NBATCH + nb)) * EMB + hh * HDIM + t] =
            (_Float16)(av / dbot_s);
    }
}

// ---------------------------------------------------------------------------
// Kernel D: output projection GEMM  out(8192x1024) = attn_h @ Wo^T + bias.
//
// TDM stages the block's 16x1024 f16 A stripe into LDS with ONE
// tensor_load_to_lds (D# built per ISA ch.8), using the TDM pad feature
// (2 dwords every 256 dwords -> effective row stride 516 dwords) so the
// per-lane ds_load_b128 fragment reads hit distinct bank groups.
// Each wave computes FOUR 16x16 tiles (64 cols), reusing one A fragment:
// per K-step = 2 ds_load_b128 + 8 global_load_b128 -> 4 v_wmma.
// ---------------------------------------------------------------------------
__global__ __launch_bounds__(256)
void outproj_wmma_kernel(const _Float16* __restrict__ A, const _Float16* __restrict__ Bw,
                         const float* __restrict__ bias, float* __restrict__ out) {
    // 16 rows * 516 dwords + 64 spare bytes = 33088 bytes
    __shared__ __align__(16) char smemA[33088];

    const int wave = threadIdx.x >> 5;
    const int lane = threadIdx.x & 31;
    const int m  = lane & 15;
    const int hi = lane >> 4;
    const int rowBase = blockIdx.x * 16;                        // 0..8176
    const int colBase = (blockIdx.y * 8 + wave) * 64;           // 0..960

    // ---- TDM: DMA A[rowBase:rowBase+16, 0:1024] (f16) into LDS ----
    if (threadIdx.x < 32) {
        unsigned lds_base = (unsigned)(uintptr_t)(void*)&smemA[0];  // LDS byte offset
        unsigned long long ga =
            (unsigned long long)(uintptr_t)(A + (size_t)rowBase * EMB);
        u32x4 g0;
        g0[0] = 1u;                                   // count=1, user descriptor
        g0[1] = lds_base;                             // lds_addr
        g0[2] = (unsigned)ga;                         // global_addr[31:0]
        g0[3] = (unsigned)((ga >> 32) & 0x01FFFFFFu)  // global_addr[56:32]
              | 0x80000000u;                          // type=2 ("image")
        u32x8 g1;
        g1[0] = 0x00010000u   // data_size=1 -> 2 bytes
              | 0x00100000u   // pad_enable
              | (7u << 22)    // pad_interval: every 256 dwords
              | (1u << 25);   // pad_amount: 2 dwords
        g1[1] = (unsigned)EMB << 16;                  // tensor_dim0 = 1024 (lo16)
        g1[2] = 16u << 16;                            // tensor_dim1 = 16   (lo16)
        g1[3] = (unsigned)EMB << 16;                  // tile_dim0 = 1024
        g1[4] = 16u;                                  // tile_dim1 = 16, tile_dim2 = 0
        g1[5] = (unsigned)EMB;                        // tensor_dim0_stride = 1024
        g1[6] = 0u;
        g1[7] = 0u;
        asm volatile("tensor_load_to_lds %0, %1" :: "s"(g0), "s"(g1) : "memory");
        __builtin_amdgcn_s_wait_tensorcnt(0);
    }
    __syncthreads();

    // Padded LDS row: 516 dwords; mid-row pad after dword 256.
    const char* ldsRow = smemA + (size_t)m * 2064;
    const _Float16* Br0 = Bw + (size_t)(colBase + m) * EMB;
    const _Float16* Br1 = Br0 + (size_t)16 * EMB;
    const _Float16* Br2 = Br0 + (size_t)32 * EMB;
    const _Float16* Br3 = Br0 + (size_t)48 * EMB;

    v8f acc[4] = {v8f{}, v8f{}, v8f{}, v8f{}};
    for (int k0 = 0; k0 < EMB; k0 += 32) {
        // A fragment from LDS (account for the 2-dword pad after dword 256)
        const int x0 = (k0 >> 1) + hi * 4;            // dword offset of chunk 0
        const int x1 = x0 + 8;                        // dword offset of chunk 1
        Frag16 fa;
        fa.h[0] = *(const v8h*)(ldsRow + 4 * (x0 + 2 * (x0 >> 8)));
        fa.h[1] = *(const v8h*)(ldsRow + 4 * (x1 + 2 * (x1 >> 8)));
        // B fragments straight from global (L2-resident, 2 MB total)
        const int kb = k0 + hi * 8;
        Frag16 fb0, fb1, fb2, fb3;
        fb0.h[0] = *(const v8h*)(Br0 + kb); fb0.h[1] = *(const v8h*)(Br0 + kb + 16);
        fb1.h[0] = *(const v8h*)(Br1 + kb); fb1.h[1] = *(const v8h*)(Br1 + kb + 16);
        fb2.h[0] = *(const v8h*)(Br2 + kb); fb2.h[1] = *(const v8h*)(Br2 + kb + 16);
        fb3.h[0] = *(const v8h*)(Br3 + kb); fb3.h[1] = *(const v8h*)(Br3 + kb + 16);
        __builtin_prefetch(Br0 + kb + 32, 0, 0);      // global_prefetch_b8
        acc[0] = __builtin_amdgcn_wmma_f32_16x16x32_f16(false, fa.v, false, fb0.v,
                                                        (short)0, acc[0], false, false);
        acc[1] = __builtin_amdgcn_wmma_f32_16x16x32_f16(false, fa.v, false, fb1.v,
                                                        (short)0, acc[1], false, false);
        acc[2] = __builtin_amdgcn_wmma_f32_16x16x32_f16(false, fa.v, false, fb2.v,
                                                        (short)0, acc[2], false, false);
        acc[3] = __builtin_amdgcn_wmma_f32_16x16x32_f16(false, fa.v, false, fb3.v,
                                                        (short)0, acc[3], false, false);
    }
#pragma unroll
    for (int tt = 0; tt < 4; ++tt) {
        const int col = colBase + tt * 16 + m;
        const float bv = bias[col];
#pragma unroll
        for (int g = 0; g < 8; ++g)
            out[(size_t)(rowBase + g + 8 * hi) * EMB + col] = acc[tt][g] + bv;
    }
}

// ---------------------------------------------------------------------------
extern "C" void kernel_launch(void* const* d_in, const int* in_sizes, int n_in,
                              void* d_out, int out_size, void* d_ws, size_t ws_size,
                              hipStream_t stream) {
    (void)in_sizes; (void)n_in; (void)out_size; (void)ws_size;
    const float* q_step = (const float*)d_in[0];
    const float* k_step = (const float*)d_in[1];
    const float* v_step = (const float*)d_in[2];
    const float* d_mem  = (const float*)d_in[3];
    const float* AV_mem = (const float*)d_in[4];
    const float* Q_mem  = (const float*)d_in[5];
    const float* K_T    = (const float*)d_in[6];
    const float* V_mem  = (const float*)d_in[7];
    const float* Wi     = (const float*)d_in[8];
    const float* bi     = (const float*)d_in[9];
    const float* Wo     = (const float*)d_in[10];
    const float* bo     = (const float*)d_in[11];
    float* out = (float*)d_out;

    // Workspace layout (16B aligned offsets)
    char* ws = (char*)d_ws;
    float*    q_s    = (float*)(ws);                                   // 64 KB
    float*    k_s    = (float*)(ws + (size_t)65536);                   // 64 KB
    float*    v_s    = (float*)(ws + (size_t)131072);                  // 64 KB
    _Float16* attn_h = (_Float16*)(ws + (size_t)196608);               // 16 MB
    _Float16* w_h    = (_Float16*)(ws + (size_t)196608 + (size_t)8192 * 1024 * 2); // 2 MB

    inproj_kernel<<<dim3(3 * EMB), 512, 0, stream>>>(q_step, k_step, v_step, Wi, bi,
                                                     q_s, k_s, v_s);
    cvt_f16_kernel<<<dim3(1024), 256, 0, stream>>>(Wo, w_h, EMB * EMB);
    attn_update_kernel<<<dim3(NBH), 256, 0, stream>>>(d_mem, AV_mem, Q_mem, K_T, V_mem,
                                                      q_s, k_s, v_s, attn_h);
    outproj_wmma_kernel<<<dim3(NSEQ, 2), 256, 0, stream>>>(attn_h, w_h, bo, out);
}
